// PFNet_52613349376210
// MI455X (gfx1250) — matrix operand order repaired and checked by reading, not compiled, not run on hardware
//
#include <hip/hip_runtime.h>
#include <hip/hip_bf16.h>
#include <math.h>
#include <stdint.h>

// ---------------- problem constants (from reference) ----------------
#define BATCH   4
#define NPTS    12800
#define FEAT    16
#define NUMCLS  8
#define DIN     23        // 8 one-hot + 15 features
#define DINP    32        // padded K for WMMA
#define DDIM    128
#define HID     256
#define BIN_SZ  640
#define NBINS   20        // NPTS / BIN_SZ
#define KNN     16
#define EDIM    47        // 2*DIN + 1
#define EDIMP   64        // padded K
#define EPB     (NPTS*KNN)        // 204800 edges per batch
#define ETOT    (BATCH*EPB)       // 819200
#define MROWS   (BATCH*NPTS)      // 51200

typedef __bf16 bf16;
typedef __attribute__((ext_vector_type(16))) __bf16 v16bf;
typedef __attribute__((ext_vector_type(8)))  __bf16 v8bf;
typedef __attribute__((ext_vector_type(8)))  float  v8f;
typedef __attribute__((ext_vector_type(4)))  unsigned v4u;
typedef __attribute__((ext_vector_type(4)))  int    v4i;
typedef __attribute__((ext_vector_type(8)))  int    v8i;

#if __has_builtin(__builtin_amdgcn_tensor_load_to_lds)
#define HAVE_TDM 1
#else
#define HAVE_TDM 0
#endif

// ---------------- helpers ----------------
__device__ __forceinline__ bf16  f2bf(float f) { return (bf16)f; }
__device__ __forceinline__ float bf2f(bf16 h)  { return (float)h; }

__device__ __forceinline__ v8f zero8() {
  v8f z;
#pragma unroll
  for (int i = 0; i < 8; ++i) z[i] = 0.0f;
  return z;
}

__device__ __forceinline__ v8f wmma_bf16(v16bf a, v16bf b, v8f c) {
  return __builtin_amdgcn_wmma_f32_16x16x32_bf16(
      false, a, false, b, (short)0, c, false, false);
}

__device__ __forceinline__ v16bf cat8(v8bf x, v8bf y) {
  return __builtin_shufflevector(x, y, 0,1,2,3,4,5,6,7,8,9,10,11,12,13,14,15);
}

// branchless activations (no exec-mask cascades)
__device__ __forceinline__ float eluf(float x)  { return fmaxf(x, 0.f) + __expf(fminf(x, 0.f)) - 1.f; }
__device__ __forceinline__ float seluf(float x) {
  const float sc = 1.0507009873554805f, al = 1.6732632423543772f;
  return sc * (fmaxf(x, 0.f) + al * (__expf(fminf(x, 0.f)) - 1.f));
}
__device__ __forceinline__ float sigm(float x) { return 1.f / (1.f + __expf(-x)); }

// ---------------- utility kernels ----------------
__global__ void k_zero_f32(float* p, long n) {
  long i = (long)blockIdx.x * blockDim.x + threadIdx.x;
  if (i < n) p[i] = 0.0f;
}

// pack f32 W[rows x cols] (logical K x N, zero padded) into WMMA B-fragment-major bf16:
// dst[(((n0t*KT + k0t)*32 + lane)*16 + i)] = W[k0t*32 + hi*16 + i][n0t*16 + l]
__global__ void k_pack_w(const float* __restrict__ src, bf16* __restrict__ dst,
                         int rows, int cols, int K, int N) {
  long tid = (long)blockIdx.x * blockDim.x + threadIdx.x;
  long tot = (long)K * N;
  if (tid >= tot) return;
  int i    = (int)(tid & 15);
  int lane = (int)((tid >> 4) & 31);
  long rest = tid >> 9;
  int KT  = K >> 5;
  int k0t = (int)(rest % KT);
  int n0t = (int)(rest / KT);
  int l = lane & 15, hi = lane >> 4;
  int sk = k0t * 32 + hi * 16 + i;
  int sn = n0t * 16 + l;
  float v = (sk < rows && sn < cols) ? src[(long)sk * cols + sn] : 0.0f;
  dst[tid] = f2bf(v);
}

// h0 = [one_hot(class,8), X[...,1:16]] padded to 32, bf16
__global__ void k_build_h0(const float* __restrict__ X, bf16* __restrict__ h0b) {
  long i = (long)blockIdx.x * blockDim.x + threadIdx.x;
  if (i >= (long)MROWS) return;
  const float* x = X + i * FEAT;
  int id = (int)x[0];
  bf16* o = h0b + i * DINP;
#pragma unroll
  for (int j = 0; j < DINP; ++j) {
    float v;
    if (j < NUMCLS)      v = (j == id) ? 1.0f : 0.0f;
    else if (j < DIN)    v = x[j - (NUMCLS - 1)];
    else                 v = 0.0f;
    o[j] = f2bf(v);
  }
}

// ---------------- generic WMMA GEMM:  out = act(A @ W + bias) ----------------
// A: MxK bf16 row-major (K%32==0), Wp: packed fragment-major (k_pack_w), N%64==0
template <int ACT, bool OF, bool OB>
__global__ __launch_bounds__(256) void k_gemm(
    const bf16* __restrict__ A, const bf16* __restrict__ Wp,
    const float* __restrict__ bias, float* __restrict__ outF,
    bf16* __restrict__ outB, int M, int K, int N) {
  const int tid = threadIdx.x, wave = tid >> 5, lane = tid & 31;
  const int l = lane & 15, hi = lane >> 4;
  const int row0 = blockIdx.x * 128 + wave * 16;
  if (row0 >= M) return;
  int arow = row0 + l; if (arow >= M) arow = M - 1;
  const bf16* Abase = A + (size_t)arow * K;
  const v16bf* W16 = (const v16bf*)Wp;
  const int KT = K >> 5;

  for (int n0 = 0; n0 < N; n0 += 64) {
    v8f acc[4];
#pragma unroll
    for (int j = 0; j < 4; ++j) acc[j] = zero8();
    for (int kt = 0; kt < KT; ++kt) {
      const bf16* ap = Abase + kt * 32;
      v8bf a0 = *(const v8bf*)(ap + hi * 16);
      v8bf a1 = *(const v8bf*)(ap + hi * 16 + 8);
      v16bf a = cat8(a0, a1);
      __builtin_prefetch(ap + 64, 0, 1);          // global_prefetch_b8
#pragma unroll
      for (int j = 0; j < 4; ++j) {
        v16bf b = W16[((size_t)(n0 / 16 + j) * KT + kt) * 32 + lane];
        acc[j] = wmma_bf16(a, b, acc[j]);
      }
    }
#pragma unroll
    for (int j = 0; j < 4; ++j) {
      const int nc = n0 + j * 16 + l;
      const float bv = bias ? bias[nc] : 0.0f;
#pragma unroll
      for (int r = 0; r < 8; ++r) {
        int row = row0 + r + 8 * hi;
        if (row < M) {
          float v = acc[j][r] + bv;
          if (ACT == 1) v = eluf(v); else if (ACT == 2) v = seluf(v);
          size_t o = (size_t)row * N + nc;
          if (OF) outF[o] = v;
          if (OB) outB[o] = f2bf(v);
        }
      }
    }
  }
}

// ---------------- LSH bin id + stable counting sort ----------------
__global__ void k_binid(const bf16* __restrict__ peb, const float* __restrict__ R,
                        int* __restrict__ binid) {
  long i = (long)blockIdx.x * blockDim.x + threadIdx.x;
  if (i >= (long)MROWS) return;
  const bf16* p = peb + i * DDIM;
  float mul[10];
#pragma unroll
  for (int j = 0; j < 10; ++j) mul[j] = 0.0f;
  for (int k = 0; k < DDIM; ++k) {
    float pv = bf2f(p[k]);
    const float* rr = R + (size_t)k * 100;
#pragma unroll
    for (int j = 0; j < 10; ++j) mul[j] += pv * rr[j];
  }
  float best = -3.0e38f; int bi = 0;
#pragma unroll
  for (int j = 0; j < NBINS; ++j) {
    float v = (j < 10) ? mul[j] : -mul[j - 10];
    if (v > best) { best = v; bi = j; }
  }
  binid[i] = bi;
}

__global__ void k_binsort(const int* __restrict__ binid, int* __restrict__ bins) {
  if (threadIdx.x != 0) return;
  const int b = blockIdx.x;
  const int* bid = binid + (long)b * NPTS;
  int* out = bins + (long)b * NPTS;
  int cnt[NBINS], off[NBINS];
  for (int i = 0; i < NBINS; ++i) cnt[i] = 0;
  for (int n = 0; n < NPTS; ++n) cnt[bid[n]]++;
  int acc = 0;
  for (int i = 0; i < NBINS; ++i) { off[i] = acc; acc += cnt[i]; }
  for (int n = 0; n < NPTS; ++n) { int id = bid[n]; out[off[id]++] = n; }
}

// ---------------- per-bin similarity (WMMA + TDM gather) + top-K ----------------
__global__ __launch_bounds__(256) void k_dm_topk(
    const bf16* __restrict__ peb, const int* __restrict__ bins,
    int* __restrict__ srcA, int* __restrict__ dstA, float* __restrict__ dval) {
  const int b = blockIdx.x / NBINS, bin = blockIdx.x % NBINS;
  const bf16* pe = peb + (size_t)b * NPTS * DDIM;
  const int* binp = bins + (size_t)b * NPTS + bin * BIN_SZ;
  int*   srcp = srcA + (size_t)b * EPB;
  int*   dstp = dstA + (size_t)b * EPB;
  float* dvp  = dval + (size_t)b * EPB;

  __shared__ bf16  At[8][16 * DDIM];   // 32 KB
  __shared__ bf16  Bt[16 * DDIM];      //  4 KB
  __shared__ float Dt[8][16 * 16];     //  8 KB
  __shared__ float topv[8][16 * KNN];  //  8 KB
  __shared__ int   topi[8][16 * KNN];  //  8 KB

  const int tid = threadIdx.x, wave = tid >> 5, lane = tid & 31;
  const int l = lane & 15, hi = lane >> 4;

  for (int rg = 0; rg < 5; ++rg) {
    const int rowBase = rg * 128 + wave * 16;
    for (int idx = lane; idx < 16 * DDIM; idx += 32) {
      int r = idx >> 7, k = idx & 127;
      int node = binp[rowBase + r];
      At[wave][idx] = pe[(size_t)node * DDIM + k];
    }
    for (int idx = lane; idx < 16 * KNN; idx += 32) {
      topv[wave][idx] = -3.0e38f; topi[wave][idx] = 0;
    }
    __syncthreads();

    for (int jc = 0; jc < BIN_SZ / 16; ++jc) {
#if HAVE_TDM
      // Tensor Data Mover gather: 16 rows (16-bit indices) x 128 bf16 -> LDS Bt
      if (wave == 0) {
        uint64_t ga = (uint64_t)(uintptr_t)pe;
        v4u g0;
        g0[0] = 1u | (1u << 31);                         // count=1, gather_mode, 16b idx
        g0[1] = (unsigned)(uintptr_t)&Bt[0];             // LDS byte address
        g0[2] = (unsigned)ga;                            // global_addr[31:0]
        g0[3] = (unsigned)((ga >> 32) & 0x1ffffffu) | (2u << 30);  // addr[56:32], type=2
        v8i g1;
        g1[0] = 1 << 16;                                 // data_size = 2 bytes
        g1[1] = (DDIM & 0xffff) << 16;                   // tensor_dim0[15:0]
        g1[2] = (NPTS & 0xffff) << 16;                   // dim0 hi=0 | tensor_dim1 lo
        g1[3] = (NPTS >> 16) | (DDIM << 16);             // tensor_dim1 hi | tile_dim0
        g1[4] = 16;                                      // tile_dim1 = #indices
        g1[5] = DDIM;                                    // tensor_dim0_stride lo
        g1[6] = 0; g1[7] = 0;
        v4i g2, g3;
#pragma unroll
        for (int q = 0; q < 4; ++q)
          g2[q] = (binp[jc * 16 + 2 * q] & 0xffff) | (binp[jc * 16 + 2 * q + 1] << 16);
#pragma unroll
        for (int q = 0; q < 4; ++q)
          g3[q] = (binp[jc * 16 + 8 + 2 * q] & 0xffff) | (binp[jc * 16 + 9 + 2 * q] << 16);
#if defined(__clang_major__) && __clang_major__ >= 23
        v8i gpad; 
#pragma unroll
        for (int q = 0; q < 8; ++q) gpad[q] = 0;
        __builtin_amdgcn_tensor_load_to_lds(g0, g1, g2, g3, gpad, 0);
#else
        __builtin_amdgcn_tensor_load_to_lds(g0, g1, g2, g3, 0);
#endif
        __builtin_amdgcn_s_wait_tensorcnt(0);
      }
      __syncthreads();
#else
      for (int idx = tid; idx < 16 * DDIM; idx += 256) {
        int c = idx >> 7, k = idx & 127;
        int node = binp[jc * 16 + c];
        Bt[idx] = pe[(size_t)node * DDIM + k];
      }
      __syncthreads();
#endif

      v8f acc = zero8();
#pragma unroll
      for (int k0 = 0; k0 < DDIM; k0 += 32) {
        v8bf a0 = *(const v8bf*)&At[wave][l * DDIM + k0 + hi * 16];
        v8bf a1 = *(const v8bf*)&At[wave][l * DDIM + k0 + hi * 16 + 8];
        v16bf bb = *(const v16bf*)&Bt[l * DDIM + k0 + hi * 16];
        acc = wmma_bf16(cat8(a0, a1), bb, acc);
      }
#pragma unroll
      for (int r = 0; r < 8; ++r) Dt[wave][(r + 8 * hi) * 16 + l] = acc[r];
      __syncthreads();

      if (lane < 16) {
        float* tv = &topv[wave][lane * KNN];
        int*   ti = &topi[wave][lane * KNN];
        for (int n = 0; n < 16; ++n) {
          float v = Dt[wave][lane * 16 + n];
          int ci = jc * 16 + n;
          if (v > tv[KNN - 1]) {
            int p = KNN - 1;
            while (p > 0 && tv[p - 1] < v) { tv[p] = tv[p - 1]; ti[p] = ti[p - 1]; --p; }
            tv[p] = v; ti[p] = ci;
          }
        }
      }
      __syncthreads();
    }

    if (lane < 16) {
      const int rlocal = rowBase + lane;
      const int srcNode = binp[rlocal];
      const size_t ebase = (size_t)(bin * BIN_SZ + rlocal) * KNN;
      const float* tv = &topv[wave][lane * KNN];
      const int*   ti = &topi[wave][lane * KNN];
#pragma unroll
      for (int k = 0; k < KNN; ++k) {
        srcp[ebase + k] = srcNode;
        dstp[ebase + k] = binp[ti[k]];
        dvp[ebase + k]  = sigm(tv[k]);
      }
    }
    __syncthreads();
  }
}

// ---------------- fused edge FFN: w = sigmoid(elu(ef@Wd1+bd1)@Wd2 + bd2) ----------------
__global__ __launch_bounds__(256) void k_edge(
    const bf16* __restrict__ h0b, const int* __restrict__ srcA, const int* __restrict__ dstA,
    const float* __restrict__ dval, const bf16* __restrict__ Wd1p,
    const float* __restrict__ bd1, const float* __restrict__ Wd2,
    const float* __restrict__ bd2, float* __restrict__ wE) {
  const int tid = threadIdx.x, wave = tid >> 5, lane = tid & 31;
  const int l = lane & 15, hi = lane >> 4;
  const long eBase = (long)blockIdx.x * 128 + wave * 16;
  if (eBase >= (long)ETOT) return;
  const int b = (int)(eBase / EPB);

  __shared__ bf16 ef[8][16 * EDIMP];

  for (int idx = lane; idx < 16 * EDIMP; idx += 32) {
    const int r = idx >> 6, c = idx & 63;
    const long e = eBase + r;
    bf16 v = f2bf(0.0f);
    const int s = srcA[e], d = dstA[e];
    if (c < DIN)            v = h0b[((size_t)b * NPTS + s) * DINP + c];
    else if (c < 2 * DIN)   v = h0b[((size_t)b * NPTS + d) * DINP + (c - DIN)];
    else if (c == 2 * DIN)  v = f2bf(dval[e]);
    ef[wave][idx] = v;
  }
  __syncthreads();

  const v16bf* W16 = (const v16bf*)Wd1p;   // packed, K=64 (KT=2), N=128 (NT=8)
  v8f acc[8];
#pragma unroll
  for (int j = 0; j < 8; ++j) acc[j] = zero8();
#pragma unroll
  for (int kt = 0; kt < 2; ++kt) {
    v8bf a0 = *(const v8bf*)&ef[wave][l * EDIMP + kt * 32 + hi * 16];
    v8bf a1 = *(const v8bf*)&ef[wave][l * EDIMP + kt * 32 + hi * 16 + 8];
    v16bf a = cat8(a0, a1);
#pragma unroll
    for (int j = 0; j < 8; ++j) {
      v16bf bb = W16[((size_t)j * 2 + kt) * 32 + lane];
      acc[j] = wmma_bf16(a, bb, acc[j]);
    }
  }

  float part[8];
#pragma unroll
  for (int r = 0; r < 8; ++r) part[r] = 0.0f;
#pragma unroll
  for (int j = 0; j < 8; ++j) {
    const float b1 = bd1[j * 16 + l];
    const float w2 = Wd2[j * 16 + l];
#pragma unroll
    for (int r = 0; r < 8; ++r) part[r] += eluf(acc[j][r] + b1) * w2;
  }
#pragma unroll
  for (int m = 1; m < 16; m <<= 1) {
#pragma unroll
    for (int r = 0; r < 8; ++r) part[r] += __shfl_xor(part[r], m, 32);
  }
  if (l == 0) {
    const float b2 = bd2[0];
#pragma unroll
    for (int r = 0; r < 8; ++r) wE[eBase + r + 8 * hi] = sigm(part[r] + b2);
  }
}

// ---------------- GHConv scatter pieces ----------------
__global__ void k_scatter_deg(const int* __restrict__ srcA, const float* __restrict__ wE,
                              float* __restrict__ deg) {
  long e = (long)blockIdx.x * blockDim.x + threadIdx.x;
  if (e >= (long)ETOT) return;
  int b = (int)(e / EPB);
  atomicAdd(&deg[(size_t)b * NPTS + srcA[e]], wE[e]);
}

__global__ void k_norm(const float* __restrict__ deg, float* __restrict__ nrm) {
  long i = (long)blockIdx.x * blockDim.x + threadIdx.x;
  if (i >= (long)MROWS) return;
  nrm[i] = rsqrtf(deg[i] + 1e-6f);
}

__global__ void k_scatter_agg(const int* __restrict__ srcA, const int* __restrict__ dstA,
                              const float* __restrict__ wE, const float* __restrict__ nrm,
                              const float* __restrict__ fhom, float* __restrict__ agg) {
  long t = (long)blockIdx.x * blockDim.x + threadIdx.x;
  if (t >= (long)ETOT * 8) return;
  long e = t >> 3; int part = (int)(t & 7);
  int b = (int)(e / EPB);
  size_t s = (size_t)b * NPTS + srcA[e];
  size_t d = (size_t)b * NPTS + dstA[e];
  float wv = wE[e] * nrm[d];
  const float* fh = fhom + d * HID;
  float* ag = agg + s * HID;
#pragma unroll
  for (int c = part * 32; c < part * 32 + 32; ++c) atomicAdd(&ag[c], wv * fh[c]);
}

__global__ void k_gh_final(const float* __restrict__ agg, const float* __restrict__ nrm,
                           const float* __restrict__ gateL, const float* __restrict__ hh,
                           bf16* __restrict__ xgB) {
  long t = (long)blockIdx.x * blockDim.x + threadIdx.x;
  if (t >= (long)MROWS * HID) return;
  long i = t >> 8;
  float a = agg[t] * nrm[i];
  float g = sigm(gateL[t]);
  xgB[t] = f2bf(seluf(g * a + (1.0f - g) * hh[t]));
}

// ---------------- output head: (256 -> 3) ----------------
__global__ void k_out(const bf16* __restrict__ xdB, const float* __restrict__ Wout,
                      const float* __restrict__ bout, float* __restrict__ out) {
  long t = (long)blockIdx.x * blockDim.x + threadIdx.x;
  if (t >= (long)MROWS * 3) return;
  long i = t / 3; int c = (int)(t % 3);
  const bf16* x = xdB + i * HID;
  float s = bout[c];
  for (int k = 0; k < HID; ++k) s += bf2f(x[k]) * Wout[k * 3 + c];
  out[t] = s;
}

// ---------------- host orchestration ----------------
static inline size_t alignUp(size_t x) { return (x + 255) & ~(size_t)255; }

extern "C" void kernel_launch(void* const* d_in, const int* in_sizes, int n_in,
                              void* d_out, int out_size, void* d_ws, size_t ws_size,
                              hipStream_t stream) {
  const float* X    = (const float*)d_in[0];
  const float* We1  = (const float*)d_in[1];  const float* be1  = (const float*)d_in[2];
  const float* We2  = (const float*)d_in[3];  const float* be2  = (const float*)d_in[4];
  const float* Wd1  = (const float*)d_in[5];  const float* bd1  = (const float*)d_in[6];
  const float* Wd2  = (const float*)d_in[7];  const float* bd2  = (const float*)d_in[8];
  const float* R    = (const float*)d_in[9];
  const float* Wenc = (const float*)d_in[10]; const float* benc = (const float*)d_in[11];
  const float* Wt   = (const float*)d_in[12]; const float* bt   = (const float*)d_in[13];
  const float* Wh   = (const float*)d_in[14];
  const float* th   = (const float*)d_in[15];
  const float* Wdec = (const float*)d_in[16]; const float* bdec = (const float*)d_in[17];
  const float* Wout = (const float*)d_in[18]; const float* bout = (const float*)d_in[19];

  char* w = (char*)d_ws; size_t off = 0;
  auto alloc = [&](size_t bytes) -> void* { void* p = w + off; off = alignUp(off + bytes); return p; };

  bf16* h0b   = (bf16*)alloc((size_t)MROWS * DINP * 2);
  bf16* t1b   = (bf16*)alloc((size_t)MROWS * DDIM * 2);
  bf16* peb   = (bf16*)alloc((size_t)MROWS * DDIM * 2);
  bf16* We1p  = (bf16*)alloc((size_t)DINP * 128 * 2);
  bf16* We2p  = (bf16*)alloc((size_t)128 * 128 * 2);
  bf16* Wd1p  = (bf16*)alloc((size_t)EDIMP * 128 * 2);
  bf16* Wencp = (bf16*)alloc((size_t)DINP * HID * 2);
  bf16* Wtp   = (bf16*)alloc((size_t)HID * HID * 2);
  bf16* Whp   = (bf16*)alloc((size_t)HID * HID * 2);
  bf16* thp   = (bf16*)alloc((size_t)HID * HID * 2);
  bf16* Wdecp = (bf16*)alloc((size_t)HID * HID * 2);
  int*  binid = (int*)alloc((size_t)MROWS * 4);
  int*  bins  = (int*)alloc((size_t)MROWS * 4);
  int*  srcA  = (int*)alloc((size_t)ETOT * 4);
  int*  dstA  = (int*)alloc((size_t)ETOT * 4);
  float* dvalA= (float*)alloc((size_t)ETOT * 4);
  float* wE   = (float*)alloc((size_t)ETOT * 4);
  float* xeF  = (float*)alloc((size_t)MROWS * HID * 4);
  bf16*  xeB  = (bf16*)alloc((size_t)MROWS * HID * 2);
  float* fhom = (float*)alloc((size_t)MROWS * HID * 4);
  float* gateL= (float*)alloc((size_t)MROWS * HID * 4);
  float* hh   = (float*)alloc((size_t)MROWS * HID * 4);
  float* deg  = (float*)alloc((size_t)MROWS * 4);
  float* nrm  = (float*)alloc((size_t)MROWS * 4);
  float* agg  = (float*)alloc((size_t)MROWS * HID * 4);
  bf16*  xgB  = (bf16*)alloc((size_t)MROWS * HID * 2);
  bf16*  xdB  = (bf16*)alloc((size_t)MROWS * HID * 2);
  (void)ws_size; (void)n_in; (void)in_sizes; (void)out_size;

  auto g1 = [](long n) { return dim3((unsigned)((n + 255) / 256)); };

  // weights -> packed bf16 fragment-major (zero padded K)
  k_pack_w<<<g1((long)DINP*128), 256, 0, stream>>>(We1, We1p, DIN, 128, DINP, 128);
  k_pack_w<<<g1((long)128*128), 256, 0, stream>>>(We2, We2p, 128, 128, 128, 128);
  k_pack_w<<<g1((long)EDIMP*128), 256, 0, stream>>>(Wd1, Wd1p, EDIM, 128, EDIMP, 128);
  k_pack_w<<<g1((long)DINP*HID), 256, 0, stream>>>(Wenc, Wencp, DIN, HID, DINP, HID);
  k_pack_w<<<g1((long)HID*HID), 256, 0, stream>>>(Wt, Wtp, HID, HID, HID, HID);
  k_pack_w<<<g1((long)HID*HID), 256, 0, stream>>>(Wh, Whp, HID, HID, HID, HID);
  k_pack_w<<<g1((long)HID*HID), 256, 0, stream>>>(th, thp, HID, HID, HID, HID);
  k_pack_w<<<g1((long)HID*HID), 256, 0, stream>>>(Wdec, Wdecp, HID, HID, HID, HID);

  // input encoding + point-embedding FFN
  k_build_h0<<<g1(MROWS), 256, 0, stream>>>(X, h0b);
  k_gemm<1,false,true><<<dim3(MROWS/128), 256, 0, stream>>>(h0b, We1p, be1, nullptr, t1b, MROWS, DINP, 128);
  k_gemm<0,false,true><<<dim3(MROWS/128), 256, 0, stream>>>(t1b, We2p, be2, nullptr, peb, MROWS, 128, 128);

  // LSH binning + per-bin kNN edges (WMMA + TDM gather)
  k_binid<<<g1(MROWS), 256, 0, stream>>>(peb, R, binid);
  k_binsort<<<dim3(BATCH), 32, 0, stream>>>(binid, bins);
  k_dm_topk<<<dim3(BATCH * NBINS), 256, 0, stream>>>(peb, bins, srcA, dstA, dvalA);

  // edge network (fused two-layer FFN)
  k_edge<<<dim3(ETOT / 128), 256, 0, stream>>>(h0b, srcA, dstA, dvalA, Wd1p, bd1, Wd2, bd2, wE);

  // encoder + GHConv matmuls
  k_gemm<2,true,true><<<dim3(MROWS/128), 256, 0, stream>>>(h0b, Wencp, benc, xeF, xeB, MROWS, DINP, HID);
  k_gemm<0,true,false><<<dim3(MROWS/128), 256, 0, stream>>>(xeB, thp, nullptr, fhom, nullptr, MROWS, HID, HID);
  k_gemm<0,true,false><<<dim3(MROWS/128), 256, 0, stream>>>(xeB, Wtp, bt, gateL, nullptr, MROWS, HID, HID);
  k_gemm<0,true,false><<<dim3(MROWS/128), 256, 0, stream>>>(xeB, Whp, nullptr, hh, nullptr, MROWS, HID, HID);

  // GHConv aggregation
  k_zero_f32<<<g1(MROWS), 256, 0, stream>>>(deg, MROWS);
  k_zero_f32<<<g1((long)MROWS*HID), 256, 0, stream>>>(agg, (long)MROWS*HID);
  k_scatter_deg<<<g1(ETOT), 256, 0, stream>>>(srcA, wE, deg);
  k_norm<<<g1(MROWS), 256, 0, stream>>>(deg, nrm);
  k_scatter_agg<<<g1((long)ETOT*8), 256, 0, stream>>>(srcA, dstA, wE, nrm, fhom, agg);
  k_gh_final<<<g1((long)MROWS*HID), 256, 0, stream>>>(agg, nrm, gateL, hh, xgB);

  // decoder + head
  k_gemm<2,false,true><<<dim3(MROWS/128), 256, 0, stream>>>(xgB, Wdecp, bdec, nullptr, xdB, MROWS, HID, HID);
  k_out<<<g1((long)MROWS*3), 256, 0, stream>>>(xdB, Wout, bout, (float*)d_out);
}
